// Model_82755429859476
// MI455X (gfx1250) — matrix operand order repaired and checked by reading, not compiled, not run on hardware
//
#include <hip/hip_runtime.h>
#include <hip/hip_bf16.h>
#include <math.h>

typedef __attribute__((ext_vector_type(16))) _Float16 v16h;
typedef __attribute__((ext_vector_type(4)))  _Float16 v4h;
typedef __attribute__((ext_vector_type(8)))  float    v8f;
typedef __attribute__((ext_vector_type(4)))  float    v4f;

#define N1 4
#define N2 8
#define PP 1024
#define DD 128

union V16U { v16h h; v4f f4[2]; };

// ---------------------------------------------------------------------------
// Kernel 1: L2-normalize rows of feats/nfeats, convert to f16.
// One wave (32 lanes) per row of 128 floats; lane holds 4 contiguous floats.
// ---------------------------------------------------------------------------
__global__ __launch_bounds__(256) void normalize_f16_kernel(
    const float* __restrict__ feats, const float* __restrict__ nfeats,
    _Float16* __restrict__ f, _Float16* __restrict__ g) {
  int wave = (blockIdx.x * blockDim.x + threadIdx.x) >> 5;   // global row id
  int lane = threadIdx.x & 31;
  const float* src;
  _Float16* dst;
  if (wave < N1 * PP) {
    src = feats + (size_t)wave * DD;
    dst = f + (size_t)wave * DD;
  } else {
    int r = wave - N1 * PP;
    src = nfeats + (size_t)r * DD;
    dst = g + (size_t)r * DD;
  }
  v4f v = *(const v4f*)(src + lane * 4);
  float ss = v.x * v.x + v.y * v.y + v.z * v.z + v.w * v.w;
#pragma unroll
  for (int off = 16; off > 0; off >>= 1)
    ss += __shfl_xor(ss, off, 32);
  float inv = rsqrtf(ss);
  v4h h;
  h.x = (_Float16)(v.x * inv);
  h.y = (_Float16)(v.y * inv);
  h.z = (_Float16)(v.z * inv);
  h.w = (_Float16)(v.w * inv);
  *(v4h*)(dst + lane * 4) = h;
}

// ---------------------------------------------------------------------------
// Kernel 2: fused GEMM + max_q.  For each (n, m, 128-row P strip):
//   maxdot[p] = max_q f[n,p,:]·g[m,q,:]   via v_wmma_f32_16x16x32_f16
// then scores_patch = sqrt(max(2-2*maxdot,0))/2 * bg[n,p].
// 8 waves/block, each owns a 16-row tile of P; B tiles (16 q-rows x 128)
// staged in LDS and shared by all 8 waves.
// ---------------------------------------------------------------------------
__global__ __launch_bounds__(256) void maxdot_kernel(
    const _Float16* __restrict__ f, const _Float16* __restrict__ g,
    const float* __restrict__ bg, float* __restrict__ spatch) {
  int b = blockIdx.x;
  int pt = b & 7; b >>= 3;        // 8 P-strips of 128 rows
  int m  = b & 7; b >>= 3;        // normal image
  int n  = b;                     // query image

  int tid  = threadIdx.x;
  int lane = tid & 31;
  int wave = tid >> 5;
  int lh   = lane & 15;
  bool hiL = (lane >= 16);

  int prow0 = pt * 128 + wave * 16;              // this wave's P-tile base

  // ---- A fragments: 16 rows x 128 K, resident for whole Q loop ----
  // ISA 16-bit A 16x32 layout: lane<16 holds K {kb..kb+7, kb+16..kb+23},
  // lane>=16 holds K {kb+8..kb+15, kb+24..kb+31}.  (row M = lane%16)
  const _Float16* fA = f + ((size_t)n * PP + prow0 + lh) * DD;
  v16h a[4];
#pragma unroll
  for (int kk = 0; kk < 4; ++kk) {
    int base = kk * 32 + (hiL ? 8 : 0);
    V16U u;
    u.f4[0] = *(const v4f*)(fA + base);          // 8 halves
    u.f4[1] = *(const v4f*)(fA + base + 16);     // 8 halves
    a[kk] = u.h;
  }

  // ---- LDS staging buffer for B tile: 16 q-rows x 128 halves = 4KB ----
  __shared__ __align__(16) _Float16 Bs[16 * DD];

  const _Float16* gM = g + (size_t)m * PP * DD;

  float rmax[8];
#pragma unroll
  for (int r = 0; r < 8; ++r) rmax[r] = -4.0f;   // dot >= -1, safe identity

  int ldr = tid >> 4;      // 0..15 : q-row within tile
  int ldc = tid & 15;      // 0..15 : 8-half chunk

  for (int qt = 0; qt < 64; ++qt) {
    __syncthreads();       // previous tile fully consumed
    *(v4f*)(Bs + ldr * DD + ldc * 8) =
        *(const v4f*)(gM + (size_t)(qt * 16 + ldr) * DD + ldc * 8);
    if (qt + 1 < 64)       // gfx1250 global_prefetch_b8 for next tile
      __builtin_prefetch(gM + (size_t)(qt + 1) * 16 * DD + tid * 8, 0, 0);
    __syncthreads();

    // ISA 16-bit B 32x16 layout: lane<16 -> col N=lane, K {kb..kb+7, kb+16..kb+23}
    //                            lane>=16 -> col N=lane-16, K {kb+8..kb+15, kb+24..kb+31}
    // i.e. same K order as A, from g row q.
    v8f acc = {};
#pragma unroll
    for (int kk = 0; kk < 4; ++kk) {
      int bbase = lh * DD + kk * 32 + (hiL ? 8 : 0);
      V16U u;
      u.f4[0] = *(const v4f*)(Bs + bbase);
      u.f4[1] = *(const v4f*)(Bs + bbase + 16);
      acc = __builtin_amdgcn_wmma_f32_16x16x32_f16(
          false, a[kk], false, u.h, (short)0, acc, false, false);
    }
#pragma unroll
    for (int r = 0; r < 8; ++r) rmax[r] = fmaxf(rmax[r], acc[r]);
  }

  // Reduce max across the 16 lanes of each half (columns of the tiles).
#pragma unroll
  for (int off = 1; off < 16; off <<= 1) {
#pragma unroll
    for (int r = 0; r < 8; ++r)
      rmax[r] = fmaxf(rmax[r], __shfl_xor(rmax[r], off, 32));
  }

  // C/D layout: lanes 0-15 VGPR r -> M=r ; lanes 16-31 VGPR r -> M=8+r.
  if (lh == 0) {
    int mbase = hiL ? 8 : 0;
    float* outrow = spatch + (size_t)(n * N2 + m) * PP;
    const float* bgrow = bg + (size_t)n * PP;
#pragma unroll
    for (int r = 0; r < 8; ++r) {
      int p = prow0 + mbase + r;
      float d = sqrtf(fmaxf(2.0f - 2.0f * rmax[r], 0.0f)) * 0.5f;
      outrow[p] = d * bgrow[p];
    }
  }
}

// ---------------------------------------------------------------------------
// Kernel 3: per-n reductions.
//   scores[n]  = mean_m max_p spatch[n,m,p]
//   spmean[n,p]= mean_m spatch[n,m,p]
// ---------------------------------------------------------------------------
__global__ __launch_bounds__(256) void reduce_kernel(
    const float* __restrict__ spatch, float* __restrict__ scores,
    float* __restrict__ spmean) {
  __shared__ float red[256];
  __shared__ float smax[N2];
  int n = blockIdx.x, tid = threadIdx.x;

  for (int m = 0; m < N2; ++m) {
    const float* row = spatch + (size_t)(n * N2 + m) * PP;
    float mx = -1e30f;
    for (int p = tid; p < PP; p += 256) mx = fmaxf(mx, row[p]);
    red[tid] = mx;
    __syncthreads();
    for (int s = 128; s > 0; s >>= 1) {
      if (tid < s) red[tid] = fmaxf(red[tid], red[tid + s]);
      __syncthreads();
    }
    if (tid == 0) smax[m] = red[0];
    __syncthreads();
  }
  if (tid == 0) {
    float s = 0.0f;
    for (int m = 0; m < N2; ++m) s += smax[m];
    scores[n] = s * (1.0f / N2);
  }
  for (int p = tid; p < PP; p += 256) {
    float s = 0.0f;
    for (int m = 0; m < N2; ++m) s += spatch[(size_t)(n * N2 + m) * PP + p];
    spmean[n * PP + p] = s * (1.0f / N2);
  }
}

// ---------------------------------------------------------------------------
// Kernel 4: bilinear upsample [4,32,32] -> [4,512,512], half-pixel centers,
// edge clamp (matches jax.image.resize 'bilinear' for pure upsampling).
// ---------------------------------------------------------------------------
__global__ __launch_bounds__(256) void upsample_kernel(
    const float* __restrict__ spmean, float* __restrict__ outpix) {
  int idx = blockIdx.x * 256 + threadIdx.x;       // 4*512*512 total
  int n   = idx >> 18;
  int rem = idx & 0x3FFFF;
  int y = rem >> 9, x = rem & 511;
  float sy = (y + 0.5f) * 0.0625f - 0.5f;
  float sx = (x + 0.5f) * 0.0625f - 0.5f;
  int y0 = (int)floorf(sy), x0 = (int)floorf(sx);
  float wy = sy - (float)y0, wx = sx - (float)x0;
  int y1 = y0 + 1 > 31 ? 31 : y0 + 1;
  int x1 = x0 + 1 > 31 ? 31 : x0 + 1;
  y0 = y0 < 0 ? 0 : y0;
  x0 = x0 < 0 ? 0 : x0;
  const float* sp = spmean + n * PP;
  float v00 = sp[y0 * 32 + x0], v01 = sp[y0 * 32 + x1];
  float v10 = sp[y1 * 32 + x0], v11 = sp[y1 * 32 + x1];
  float v0 = v00 + (v01 - v00) * wx;
  float v1 = v10 + (v11 - v10) * wx;
  outpix[idx] = v0 + (v1 - v0) * wy;
}

// ---------------------------------------------------------------------------
extern "C" void kernel_launch(void* const* d_in, const int* in_sizes, int n_in,
                              void* d_out, int out_size, void* d_ws, size_t ws_size,
                              hipStream_t stream) {
  const float* feats  = (const float*)d_in[0];   // [4,1024,128]
  const float* nfeats = (const float*)d_in[1];   // [8,1024,128]
  const float* bg     = (const float*)d_in[2];   // [4,1024]
  // d_in[3..6] = PH,PW,H,W scalars (fixed: 32,32,512,512)

  char* ws = (char*)d_ws;
  _Float16* f      = (_Float16*)ws;                                  // 1 MB
  _Float16* g      = (_Float16*)(ws + (1u << 20));                   // 2 MB
  float*    spatch = (float*)(ws + 3u * (1u << 20));                 // 128 KB
  float*    spmean = (float*)(ws + 3u * (1u << 20) + (128u << 10));  // 16 KB

  float* scores = (float*)d_out;        // [4]
  float* pixmap = scores + N1;          // [4,512,512]

  // 12288 rows, 8 waves/block of 256 threads -> 1536 blocks
  normalize_f16_kernel<<<1536, 256, 0, stream>>>(feats, nfeats, f, g);
  // 4 * 8 * 8 P-strips = 256 blocks
  maxdot_kernel<<<256, 256, 0, stream>>>(f, g, bg, spatch);
  reduce_kernel<<<N1, 256, 0, stream>>>(spatch, scores, spmean);
  upsample_kernel<<<(N1 * 512 * 512) / 256, 256, 0, stream>>>(spmean, pixmap);
}
